// MultiHeadAttention_15049565405414
// MI455X (gfx1250) — matrix operand order repaired and checked by reading, not compiled, not run on hardware
//
#include <hip/hip_runtime.h>
#include <hip/hip_bf16.h>
#include <math.h>

// ---------------------------------------------------------------------------
// MHA forward for MI455X (gfx1250): bf16 WMMA + async global->LDS copies
// ---------------------------------------------------------------------------

#define D_MODEL 1024
#define NUM_HEADS 16
#define D_K 64
#define BATCH 4
#define SEQ 2048
#define MROWS (BATCH * SEQ)          // 8192
#define KSTEP 32
#define TILE 64

typedef __attribute__((ext_vector_type(16))) __bf16 v16bf;
typedef __attribute__((ext_vector_type(8)))  float  v8f;
typedef __attribute__((ext_vector_type(4)))  float  f32x4;
typedef __attribute__((ext_vector_type(4)))  __bf16 bf16x4;

__device__ __forceinline__ v8f wmma_bf16(v16bf a, v16bf b, v8f c) {
  // 8 args: (neg_a, A, neg_b, B, c_mod, C, reuse_a, reuse_b)
  return __builtin_amdgcn_wmma_f32_16x16x32_bf16(false, a, false, b,
                                                 (short)0, c, false, false);
}

// A-fragment K index (16-bit A 16x32 ISA layout): VGPR0-3 cover K 0-7
// (lanes 0-15) / 8-15 (lanes 16-31), VGPR4-7 cover K 16-23 / 24-31.
__device__ __forceinline__ int a_kidx(int lane, int v) {
  return ((v & 4) << 2) + ((lane >> 4) << 3) + ((v & 3) << 1);
}
// B-fragment (32x16, column-major per lane): same K order as A:
// VGPR0-3 cover K 0-7 (lanes 0-15) / 8-15 (lanes 16-31), VGPR4-7 cover K 16-23 / 24-31.
__device__ __forceinline__ int b_kidx(int lane, int v) {
  return ((v & 4) << 2) + ((lane >> 4) << 3) + ((v & 3) << 1);
}

// Async 16B copy global -> LDS (per-lane), tracked by ASYNCcnt.
// LDS offset = flat shared-aperture address truncated to 32 bits (ISA 10.2).
__device__ __forceinline__ void async_copy_b128(void* lds_ptr, const void* gptr) {
  unsigned off = (unsigned)(unsigned long long)lds_ptr;
  asm volatile("global_load_async_to_lds_b128 %0, %1, off"
               :: "v"(off), "v"(gptr) : "memory");
}
__device__ __forceinline__ void wait_async() {
  asm volatile("s_wait_asynccnt 0" ::: "memory");
}

__device__ __forceinline__ bf16x4 cvt4(f32x4 x) {
  bf16x4 r;
#pragma unroll
  for (int j = 0; j < 4; ++j) r[j] = (__bf16)x[j];
  return r;
}

// ---------------------------------------------------------------------------
// Kernel 1: Out(bf16)[m,n] = sum_k X(f32)[m,k] * W(f32)[n,k] + bias[n]
// 64x64 tile / 4-wave block; wave w owns rows w*16..w*16+15.
// ---------------------------------------------------------------------------
__global__ __launch_bounds__(128)
void gemm_xwT_bias(const float* __restrict__ X, const float* __restrict__ W,
                   const float* __restrict__ bias, __bf16* __restrict__ Out,
                   int M, int N, int K) {
  __shared__ __align__(16) __bf16 lA[TILE][KSTEP + 8];
  __shared__ __align__(16) __bf16 lB[TILE][KSTEP + 8];
  const int tid  = threadIdx.x;
  const int lane = tid & 31;
  const int wave = tid >> 5;
  const int m0 = blockIdx.x * TILE;
  const int n0 = blockIdx.y * TILE;

  v8f acc[4];
  for (int t = 0; t < 4; ++t)
    for (int i = 0; i < 8; ++i) acc[t][i] = 0.0f;

  for (int k0 = 0; k0 < K; k0 += KSTEP) {
#pragma unroll
    for (int it = 0; it < 4; ++it) {            // 512 float4 chunks / 128 thr
      const int c = tid + it * 128;
      const int m = c >> 3, k4 = (c & 7) << 2;
      f32x4 xa = *(const f32x4*)&X[(size_t)(m0 + m) * K + (k0 + k4)];
      f32x4 wb = *(const f32x4*)&W[(size_t)(n0 + m) * K + (k0 + k4)];
      *(bf16x4*)&lA[m][k4] = cvt4(xa);
      *(bf16x4*)&lB[m][k4] = cvt4(wb);
      __builtin_prefetch(&X[(size_t)(m0 + m) * K + (k0 + KSTEP + k4)], 0, 1);
    }
    __syncthreads();

    v16bf af;
    const int am = (wave << 4) + (lane & 15);
#pragma unroll
    for (int v = 0; v < 8; ++v) {
      const int kb = a_kidx(lane, v);
      af[2 * v]     = lA[am][kb];
      af[2 * v + 1] = lA[am][kb + 1];
    }
#pragma unroll
    for (int t = 0; t < 4; ++t) {
      v16bf bfr;
      const int bn = (t << 4) + (lane & 15);
#pragma unroll
      for (int v = 0; v < 8; ++v) {
        const int kb = b_kidx(lane, v);
        bfr[2 * v]     = lB[bn][kb];
        bfr[2 * v + 1] = lB[bn][kb + 1];
      }
      acc[t] = wmma_bf16(af, bfr, acc[t]);
    }
    __syncthreads();
  }

#pragma unroll
  for (int t = 0; t < 4; ++t) {
    const int n = n0 + (t << 4) + (lane & 15);
    const float bv = bias[n];
#pragma unroll
    for (int r = 0; r < 8; ++r) {
      const int m = m0 + (wave << 4) + r + ((lane >> 4) << 3);
      Out[(size_t)m * N + n] = (__bf16)(acc[t][r] + bv);
    }
  }
}

// ---------------------------------------------------------------------------
// Kernel 1b: same GEMM but X already bf16 (async-copied), output fp32.
// ---------------------------------------------------------------------------
__global__ __launch_bounds__(128)
void gemm_bf16_out(const __bf16* __restrict__ X, const float* __restrict__ W,
                   const float* __restrict__ bias, float* __restrict__ Out,
                   int M, int N, int K) {
  __shared__ __align__(16) __bf16 lA[TILE][KSTEP + 8];
  __shared__ __align__(16) __bf16 lB[TILE][KSTEP + 8];
  const int tid  = threadIdx.x;
  const int lane = tid & 31;
  const int wave = tid >> 5;
  const int m0 = blockIdx.x * TILE;
  const int n0 = blockIdx.y * TILE;

  v8f acc[4];
  for (int t = 0; t < 4; ++t)
    for (int i = 0; i < 8; ++i) acc[t][i] = 0.0f;

  for (int k0 = 0; k0 < K; k0 += KSTEP) {
#pragma unroll
    for (int it = 0; it < 2; ++it) {            // A: 256 b128 chunks / 128 thr
      const int c = tid + it * 128;
      const int m = c >> 2, d8 = (c & 3) << 3;
      async_copy_b128(&lA[m][d8], &X[(size_t)(m0 + m) * K + (k0 + d8)]);
    }
#pragma unroll
    for (int it = 0; it < 4; ++it) {            // B: fp32 -> bf16 convert
      const int c = tid + it * 128;
      const int m = c >> 3, k4 = (c & 7) << 2;
      f32x4 wb = *(const f32x4*)&W[(size_t)(n0 + m) * K + (k0 + k4)];
      *(bf16x4*)&lB[m][k4] = cvt4(wb);
    }
    wait_async();
    __syncthreads();

    v16bf af;
    const int am = (wave << 4) + (lane & 15);
#pragma unroll
    for (int v = 0; v < 8; ++v) {
      const int kb = a_kidx(lane, v);
      af[2 * v]     = lA[am][kb];
      af[2 * v + 1] = lA[am][kb + 1];
    }
#pragma unroll
    for (int t = 0; t < 4; ++t) {
      v16bf bfr;
      const int bn = (t << 4) + (lane & 15);
#pragma unroll
      for (int v = 0; v < 8; ++v) {
        const int kb = b_kidx(lane, v);
        bfr[2 * v]     = lB[bn][kb];
        bfr[2 * v + 1] = lB[bn][kb + 1];
      }
      acc[t] = wmma_bf16(af, bfr, acc[t]);
    }
    __syncthreads();
  }

#pragma unroll
  for (int t = 0; t < 4; ++t) {
    const int n = n0 + (t << 4) + (lane & 15);
    const float bv = bias[n];
#pragma unroll
    for (int r = 0; r < 8; ++r) {
      const int m = m0 + (wave << 4) + r + ((lane >> 4) << 3);
      Out[(size_t)m * N + n] = acc[t][r] + bv;
    }
  }
}

// ---------------------------------------------------------------------------
// Kernel 2: raw scores[b,h,q,k] = scale * dot(Q[b,q,h,:], K[b,k,h,:])  (Dk=64)
// grid: (SEQ/64, SEQ/64, B*H), block 128. Q/K tiles async-copied (bf16).
// ---------------------------------------------------------------------------
__global__ __launch_bounds__(128)
void attn_scores(const __bf16* __restrict__ Qp, const __bf16* __restrict__ Kp,
                 float* __restrict__ attn, float scale) {
  __shared__ __align__(16) __bf16 lQ[TILE][D_K + 8];
  __shared__ __align__(16) __bf16 lK[TILE][D_K + 8];
  const int tid  = threadIdx.x;
  const int lane = tid & 31;
  const int wave = tid >> 5;
  const int bh = blockIdx.z;
  const int b  = bh >> 4;
  const int h  = bh & 15;
  const int q0 = blockIdx.x * TILE;
  const int c0 = blockIdx.y * TILE;

#pragma unroll
  for (int it = 0; it < 4; ++it) {              // 512 b128 chunks each
    const int c = tid + it * 128;
    const int r = c >> 3, d8 = (c & 7) << 3;
    async_copy_b128(&lQ[r][d8],
                    &Qp[(size_t)(b * SEQ + q0 + r) * D_MODEL + h * D_K + d8]);
    async_copy_b128(&lK[r][d8],
                    &Kp[(size_t)(b * SEQ + c0 + r) * D_MODEL + h * D_K + d8]);
  }
  wait_async();
  __syncthreads();

  v8f acc[4];
  for (int t = 0; t < 4; ++t)
    for (int i = 0; i < 8; ++i) acc[t][i] = 0.0f;

#pragma unroll
  for (int kk = 0; kk < D_K; kk += KSTEP) {
    v16bf af;
    const int am = (wave << 4) + (lane & 15);
#pragma unroll
    for (int v = 0; v < 8; ++v) {
      const int kb = kk + a_kidx(lane, v);
      af[2 * v]     = lQ[am][kb];
      af[2 * v + 1] = lQ[am][kb + 1];
    }
#pragma unroll
    for (int t = 0; t < 4; ++t) {
      v16bf bfr;
      const int bn = (t << 4) + (lane & 15);    // key index n
#pragma unroll
      for (int v = 0; v < 8; ++v) {
        const int kb = kk + b_kidx(lane, v);    // dim index k; B[k][n]=K[n][k]
        bfr[2 * v]     = lK[bn][kb];
        bfr[2 * v + 1] = lK[bn][kb + 1];
      }
      acc[t] = wmma_bf16(af, bfr, acc[t]);
    }
  }

#pragma unroll
  for (int t = 0; t < 4; ++t) {
    const int n = c0 + (t << 4) + (lane & 15);
#pragma unroll
    for (int r = 0; r < 8; ++r) {
      const int q = q0 + (wave << 4) + r + ((lane >> 4) << 3);
      attn[((size_t)bh * SEQ + q) * SEQ + n] = acc[t][r] * scale;
    }
  }
}

// ---------------------------------------------------------------------------
// Kernel 3: in-place softmax over each 2048-wide row. One block per row.
// ---------------------------------------------------------------------------
__global__ __launch_bounds__(256)
void softmax_inplace(float* __restrict__ attn) {
  float* row = attn + (size_t)blockIdx.x * SEQ;
  __shared__ float red[256];
  const int t = threadIdx.x;
  float vals[SEQ / 256];
  float mx = -INFINITY;
#pragma unroll
  for (int i = 0; i < SEQ / 256; ++i) {
    vals[i] = row[t + i * 256];
    mx = fmaxf(mx, vals[i]);
  }
  red[t] = mx; __syncthreads();
  for (int s = 128; s > 0; s >>= 1) {
    if (t < s) red[t] = fmaxf(red[t], red[t + s]);
    __syncthreads();
  }
  mx = red[0]; __syncthreads();

  float sum = 0.0f;
#pragma unroll
  for (int i = 0; i < SEQ / 256; ++i) {
    vals[i] = __expf(vals[i] - mx);
    sum += vals[i];
  }
  red[t] = sum; __syncthreads();
  for (int s = 128; s > 0; s >>= 1) {
    if (t < s) red[t] += red[t + s];
    __syncthreads();
  }
  const float inv = 1.0f / red[0];
#pragma unroll
  for (int i = 0; i < SEQ / 256; ++i) row[t + i * 256] = vals[i] * inv;
}

// ---------------------------------------------------------------------------
// Kernel 4: attn_out[b,q,h*64+n] = sum_k P[b,h,q,k] * V[b,k,h*64+n]
// grid: (SEQ/64, 1, B*H). P fp32->bf16 vectorized; V tile async-copied.
// ---------------------------------------------------------------------------
__global__ __launch_bounds__(128)
void attn_pv(const float* __restrict__ attn, const __bf16* __restrict__ Vp,
             __bf16* __restrict__ AOut) {
  __shared__ __align__(16) __bf16 lP[TILE][KSTEP + 8];
  __shared__ __align__(16) __bf16 lV[KSTEP][D_K + 8];
  const int tid  = threadIdx.x;
  const int lane = tid & 31;
  const int wave = tid >> 5;
  const int bh = blockIdx.z;
  const int b  = bh >> 4;
  const int h  = bh & 15;
  const int q0 = blockIdx.x * TILE;
  const float* Prow = attn + ((size_t)bh * SEQ + q0) * SEQ;

  v8f acc[4];
  for (int t = 0; t < 4; ++t)
    for (int i = 0; i < 8; ++i) acc[t][i] = 0.0f;

  for (int k0 = 0; k0 < SEQ; k0 += KSTEP) {
#pragma unroll
    for (int it = 0; it < 2; ++it) {            // V: 256 b128 chunks
      const int c = tid + it * 128;
      const int r = c >> 3, d8 = (c & 7) << 3;
      async_copy_b128(&lV[r][d8],
                      &Vp[(size_t)(b * SEQ + k0 + r) * D_MODEL + h * D_K + d8]);
    }
#pragma unroll
    for (int it = 0; it < 4; ++it) {            // P: fp32 -> bf16
      const int c = tid + it * 128;
      const int m = c >> 3, k4 = (c & 7) << 2;
      f32x4 p = *(const f32x4*)&Prow[(size_t)m * SEQ + (k0 + k4)];
      *(bf16x4*)&lP[m][k4] = cvt4(p);
      __builtin_prefetch(&Prow[(size_t)m * SEQ + (k0 + KSTEP + k4)], 0, 1);
    }
    wait_async();
    __syncthreads();

    v16bf af;
    const int am = (wave << 4) + (lane & 15);
#pragma unroll
    for (int v = 0; v < 8; ++v) {
      const int kb = a_kidx(lane, v);
      af[2 * v]     = lP[am][kb];
      af[2 * v + 1] = lP[am][kb + 1];
    }
#pragma unroll
    for (int t = 0; t < 4; ++t) {
      v16bf bfr;
      const int bn = (t << 4) + (lane & 15);    // output dim n
#pragma unroll
      for (int v = 0; v < 8; ++v) {
        const int kb = b_kidx(lane, v);         // B[k][n] = V[k][n]
        bfr[2 * v]     = lV[kb][bn];
        bfr[2 * v + 1] = lV[kb + 1][bn];
      }
      acc[t] = wmma_bf16(af, bfr, acc[t]);
    }
    __syncthreads();
  }

#pragma unroll
  for (int t = 0; t < 4; ++t) {
    const int n = (t << 4) + (lane & 15);
#pragma unroll
    for (int r = 0; r < 8; ++r) {
      const int q = q0 + (wave << 4) + r + ((lane >> 4) << 3);
      AOut[(size_t)(b * SEQ + q) * D_MODEL + h * D_K + n] = (__bf16)acc[t][r];
    }
  }
}

// ---------------------------------------------------------------------------
// Host launcher
// ---------------------------------------------------------------------------
extern "C" void kernel_launch(void* const* d_in, const int* in_sizes, int n_in,
                              void* d_out, int out_size, void* d_ws, size_t ws_size,
                              hipStream_t stream) {
  (void)in_sizes; (void)n_in; (void)out_size; (void)ws_size;
  const float* q  = (const float*)d_in[0];
  const float* k  = (const float*)d_in[1];
  const float* v  = (const float*)d_in[2];
  const float* Wq = (const float*)d_in[3];
  const float* bq = (const float*)d_in[4];
  const float* Wk = (const float*)d_in[5];
  const float* bk = (const float*)d_in[6];
  const float* Wv = (const float*)d_in[7];
  const float* bv = (const float*)d_in[8];
  const float* Wo = (const float*)d_in[9];
  const float* bo = (const float*)d_in[10];

  float* out  = (float*)d_out;                                   // [B,S,D]
  float* attn = out + (size_t)BATCH * SEQ * D_MODEL;             // [B,H,S,S]

  const size_t elems = (size_t)MROWS * D_MODEL;                  // 8Mi
  __bf16* Qp = (__bf16*)d_ws;
  __bf16* Kp = Qp + elems;
  __bf16* Vp = Kp + elems;
  __bf16* AO = Vp + elems;                                       // 64 MB total

  dim3 gp(MROWS / TILE, D_MODEL / TILE);                         // 128 x 16
  gemm_xwT_bias<<<gp, 128, 0, stream>>>(q, Wq, bq, Qp, MROWS, D_MODEL, D_MODEL);
  gemm_xwT_bias<<<gp, 128, 0, stream>>>(k, Wk, bk, Kp, MROWS, D_MODEL, D_MODEL);
  gemm_xwT_bias<<<gp, 128, 0, stream>>>(v, Wv, bv, Vp, MROWS, D_MODEL, D_MODEL);

  dim3 gs(SEQ / TILE, SEQ / TILE, BATCH * NUM_HEADS);            // 32 x 32 x 64
  attn_scores<<<gs, 128, 0, stream>>>(Qp, Kp, attn, 0.125f);     // 1/sqrt(64)

  softmax_inplace<<<BATCH * NUM_HEADS * SEQ, 256, 0, stream>>>(attn);

  dim3 gpv(SEQ / TILE, 1, BATCH * NUM_HEADS);                    // 32 x 1 x 64
  attn_pv<<<gpv, 128, 0, stream>>>(attn, Vp, AO);

  gemm_bf16_out<<<gp, 128, 0, stream>>>(AO, Wo, bo, out, MROWS, D_MODEL, D_MODEL);
}
